// AdaptiveSparseAttention_57672820850753
// MI455X (gfx1250) — matrix operand order repaired and checked by reading, not compiled, not run on hardware
//
#include <hip/hip_runtime.h>
#include <hip/hip_bf16.h>

// ---------------------------------------------------------------------------
// Types for CDNA5 WMMA (wave32): D(16x16,f32) = A(16x32,bf16) x B(32x16,bf16) + C
// ---------------------------------------------------------------------------
typedef __attribute__((ext_vector_type(16))) __bf16 v16bf;
typedef __attribute__((ext_vector_type(8)))  __bf16 v8bf;
typedef __attribute__((ext_vector_type(8)))  float  v8f;

#define DEV __device__ __forceinline__

DEV unsigned short f2bfu(float f) {            // f32 -> bf16 bits, round-to-nearest-even
  unsigned u = __builtin_bit_cast(unsigned, f);
  u += 0x7FFFu + ((u >> 16) & 1u);
  return (unsigned short)(u >> 16);
}

DEV v8f wmma_bf16(v16bf a, v16bf b, v8f c) {
  // 8-arg form: (neg_a, A, neg_b, B, c_mod, C, reuse_a, reuse_b)
  return __builtin_amdgcn_wmma_f32_16x16x32_bf16(false, a, false, b, (short)0, c,
                                                 false, false);
}

DEV v8bf ld8(const unsigned short* p) { return *(const v8bf*)p; }  // 16B-aligned b128

DEV v16bf cat16(v8bf lo, v8bf hi) {
  return __builtin_shufflevector(lo, hi, 0, 1, 2, 3, 4, 5, 6, 7,
                                 8, 9, 10, 11, 12, 13, 14, 15);
}

// ---------------------------------------------------------------------------
// Fragment loaders per ISA 7.12.2 (wave32) layouts — all-vector (b128) loads.
// A (16x32 bf16): lane L: row=L%16; Kbase=(L>=16)?8:0; runs [Kb..Kb+7],[Kb+16..Kb+23]
// B (32x16 bf16): lane L: col=L%16; Kbase=(L>=16)?8:0; runs [Kb..Kb+7],[Kb+16..Kb+23]
// C/D (16x16 f32): vgpr r: row = r + (L>=16?8:0); col = L%16
// ---------------------------------------------------------------------------
DEV v16bf load_afrag(const unsigned short* p, int ld, int row0, int k0) {
  int lane = threadIdx.x & 31;
  const unsigned short* q =
      p + (size_t)(row0 + (lane & 15)) * ld + k0 + ((lane >> 4) << 3);
  return cat16(ld8(q), ld8(q + 16));
}

// B = M^T with M row-major (rows of M are columns of B) -> contiguous row reads
DEV v16bf load_bfragT(const unsigned short* p, int ld, int k0, int n0) {
  int lane = threadIdx.x & 31;
  const unsigned short* q =
      p + (size_t)(n0 + (lane & 15)) * ld + k0 + ((lane >> 4) << 3);
  return cat16(ld8(q), ld8(q + 16));
}

// ---------------------------------------------------------------------------
// Problem constants
// ---------------------------------------------------------------------------
#define SB     2048            // sequence length
#define HH     2048            // hidden
#define NH     16              // heads
#define HD     128             // head dim
#define BB     2               // batch
#define NTOT   (BB * SB * HH)  // 8388608 elements in hidden_states

// ---------------------------------------------------------------------------
// Kernel 0: f32 -> bf16 cast, 4 elements/thread (memory-bound one-pass)
// ---------------------------------------------------------------------------
__global__ void k_cast(const float* __restrict__ src,
                       unsigned short* __restrict__ dst, unsigned int n4) {
  unsigned int i = blockIdx.x * blockDim.x + threadIdx.x;
  unsigned int stride = gridDim.x * blockDim.x;
  for (; i < n4; i += stride) {
    float4 f = ((const float4*)src)[i];
    ushort4 u;
    u.x = f2bfu(f.x); u.y = f2bfu(f.y); u.z = f2bfu(f.z); u.w = f2bfu(f.w);
    ((ushort4*)dst)[i] = u;
  }
}

// ---------------------------------------------------------------------------
// Kernel 1: activation density count  (deterministic: unsigned sum)
// ---------------------------------------------------------------------------
__global__ void k_density(const float* __restrict__ x, unsigned int n,
                          unsigned int* __restrict__ cnt) {
  unsigned int i = blockIdx.x * blockDim.x + threadIdx.x;
  unsigned int stride = gridDim.x * blockDim.x;
  unsigned int c = 0;
  for (; i < n; i += stride) c += (__builtin_fabsf(x[i]) > 0.1f) ? 1u : 0u;
#pragma unroll
  for (int m = 16; m >= 1; m >>= 1) c += __shfl_xor(c, m, 32);
  if ((threadIdx.x & 31) == 0) atomicAdd(cnt, c);
}

__global__ void k_flag(const unsigned int* __restrict__ cnt, int* __restrict__ flag,
                       unsigned int n) {
  if (threadIdx.x == 0)
    *flag = (2ull * (unsigned long long)(*cnt) > (unsigned long long)n) ? 1 : 0;
}

// ---------------------------------------------------------------------------
// Kernel 2: QKV projections  Y = X @ W^T + b  (bf16 in, bf16 out)
// grid (64, 32, 3), block 128 (4 waves). Wave w: 16 rows x 64 cols.
// Q,K stored row-major (B*S, H); V stored TRANSPOSED (B, H, S) so the
// attention kernel's V fragments are contiguous b128 loads.
// ---------------------------------------------------------------------------
__global__ void k_qkv(const unsigned short* __restrict__ Xb,
                      const unsigned short* __restrict__ Wqb, const float* __restrict__ bq,
                      const unsigned short* __restrict__ Wkb, const float* __restrict__ bk,
                      const unsigned short* __restrict__ Wvb, const float* __restrict__ bv,
                      unsigned short* __restrict__ Qb,
                      unsigned short* __restrict__ Kb,
                      unsigned short* __restrict__ Vt) {
  const unsigned short* W;
  const float* bias;
  if (blockIdx.z == 0)      { W = Wqb; bias = bq; }
  else if (blockIdx.z == 1) { W = Wkb; bias = bk; }
  else                      { W = Wvb; bias = bv; }

  int wave = threadIdx.x >> 5;
  int lane = threadIdx.x & 31;
  int row0  = (blockIdx.x * 4 + wave) * 16;
  int ncol0 = blockIdx.y * 64;

  v8f acc[4] = {v8f{}, v8f{}, v8f{}, v8f{}};
  for (int k0 = 0; k0 < HH; k0 += 32) {
    v16bf a = load_afrag(Xb, HH, row0, k0);
#pragma unroll
    for (int j = 0; j < 4; ++j) {
      v16bf b = load_bfragT(W, HH, k0, ncol0 + j * 16);
      acc[j] = wmma_bf16(a, b, acc[j]);
    }
  }

  int half = lane >> 4;
  if (blockIdx.z == 2) {  // V: transposed store (B, H, S)
#pragma unroll
    for (int j = 0; j < 4; ++j) {
      int col = ncol0 + j * 16 + (lane & 15);
      float bb = bias[col];
#pragma unroll
      for (int r = 0; r < 8; ++r) {
        int row = row0 + r + half * 8;       // row = b*S + s
        int bi = row >> 11, s = row & (SB - 1);
        Vt[((size_t)bi * HH + col) * SB + s] = f2bfu(acc[j][r] + bb);
      }
    }
  } else {
    unsigned short* out = (blockIdx.z == 0) ? Qb : Kb;
#pragma unroll
    for (int j = 0; j < 4; ++j) {
      int col = ncol0 + j * 16 + (lane & 15);
      float bb = bias[col];
#pragma unroll
      for (int r = 0; r < 8; ++r) {
        int row = row0 + r + half * 8;
        out[(size_t)row * HH + col] = f2bfu(acc[j][r] + bb);
      }
    }
  }
}

// ---------------------------------------------------------------------------
// Mask: local window / strided / global token 0
// ---------------------------------------------------------------------------
DEV bool maskf(int q, int k, int local) {
  if (q == 0 || k == 0) return true;
  int d = q - k;
  if (local) { int ad = d < 0 ? -d : d; return ad <= 64; }
  return (d & 3) == 0;  // |d| % 4 == 0  <=>  d % 4 == 0
}

// ---------------------------------------------------------------------------
// Kernel 3: flash attention per (query-tile of 16, head, batch)
// grid (32, 16, 2), block 128 (4 waves, wave w -> qtile bx*4+w)
// 64-key tiles: one shuffle-reduce + one O-rescale per 64 keys (softmax VALU
// amortized 2x vs 32-key tiles). Local mode: wave-uniform tile skipping.
// ---------------------------------------------------------------------------
__global__ void k_attn(const unsigned short* __restrict__ Q,
                       const unsigned short* __restrict__ K,
                       const unsigned short* __restrict__ Vt,
                       const int* __restrict__ flagp,
                       unsigned short* __restrict__ O) {
  __shared__ __align__(16) unsigned short ldsP[4][16 * 64];

  int wave = threadIdx.x >> 5;
  int lane = threadIdx.x & 31;
  int half = lane >> 4;
  int ln   = lane & 15;

  int qt = blockIdx.x * 4 + wave;     // 0..127
  int h  = blockIdx.y;                // 0..15
  int b  = blockIdx.z;                // 0..1
  int bS = b * SB;
  int c0 = h * HD;
  int q0 = qt * 16;
  int local = *flagp;                 // wave-uniform

  // Q fragments for the whole 128-wide head dim (4 x K32), loaded once
  v16bf aQ[4];
#pragma unroll
  for (int kk = 0; kk < 4; ++kk)
    aQ[kk] = load_afrag(Q + (size_t)bS * HH + c0, HH, q0, kk * 32);

  v8f o[8] = {v8f{}, v8f{}, v8f{}, v8f{}, v8f{}, v8f{}, v8f{}, v8f{}};
  float m_[8], l_[8];
#pragma unroll
  for (int r = 0; r < 8; ++r) { m_[r] = -1e30f; l_[r] = 0.0f; }

  const float scale = 0.08838834764831845f;  // 1/(sqrt(128)*TEMP)
  const unsigned short* Kbase = K + (size_t)bS * HH + c0;
  const unsigned short* Vbase = Vt + (size_t)b * HH * SB;  // (H, S) for batch b

  for (int kb0 = 0; kb0 < SB; kb0 += 64) {
    // ---- wave-uniform tile skip (local window + global tile 0) ----
    if (qt != 0 && kb0 != 0 && local) {
      if (kb0 > q0 + 15 + 64 || kb0 + 63 < q0 - 64) continue;
    }

    // ---- scores: four 16x16 tiles over 64 keys (16 WMMAs) ----
    v8f s[4] = {v8f{}, v8f{}, v8f{}, v8f{}};
#pragma unroll
    for (int kk = 0; kk < 4; ++kk) {
#pragma unroll
      for (int t = 0; t < 4; ++t) {
        v16bf bK = load_bfragT(Kbase, HH, kk * 32, kb0 + t * 16);
        s[t] = wmma_bf16(aQ[kk], bK, s[t]);
      }
    }

    // ---- scale + mask ----
    float p[4][8];
#pragma unroll
    for (int t = 0; t < 4; ++t)
#pragma unroll
      for (int r = 0; r < 8; ++r) {
        int qi = q0 + r + half * 8;
        p[t][r] = maskf(qi, kb0 + t * 16 + ln, local) ? s[t][r] * scale : -1e30f;
      }

    // ---- online softmax: one reduce + one rescale per 64 keys ----
#pragma unroll
    for (int r = 0; r < 8; ++r) {
      float mx = fmaxf(fmaxf(p[0][r], p[1][r]), fmaxf(p[2][r], p[3][r]));
#pragma unroll
      for (int d = 1; d < 16; d <<= 1) mx = fmaxf(mx, __shfl_xor(mx, d, 32));
      float mnew  = fmaxf(m_[r], mx);
      float alpha = __expf(m_[r] - mnew);
      float sum = 0.0f;
#pragma unroll
      for (int t = 0; t < 4; ++t) {
        p[t][r] = __expf(p[t][r] - mnew);
        sum += p[t][r];
      }
#pragma unroll
      for (int d = 1; d < 16; d <<= 1) sum += __shfl_xor(sum, d, 32);
      l_[r] = l_[r] * alpha + sum;
      m_[r] = mnew;
#pragma unroll
      for (int j = 0; j < 8; ++j) o[j][r] *= alpha;
    }

    // ---- transpose P (C layout -> A layout) through LDS as bf16 ----
#pragma unroll
    for (int t = 0; t < 4; ++t)
#pragma unroll
      for (int r = 0; r < 8; ++r)
        ldsP[wave][(r + half * 8) * 64 + t * 16 + ln] = f2bfu(p[t][r]);
    asm volatile("s_wait_dscnt 0" ::: "memory");  // same-wave LDS RAW

    v16bf aP0, aP1;
    {
      const unsigned short* base = &ldsP[wave][ln * 64 + (half << 3)];
      aP0 = cat16(*(const v8bf*)(base),      *(const v8bf*)(base + 16));
      aP1 = cat16(*(const v8bf*)(base + 32), *(const v8bf*)(base + 48));
    }

    // ---- O(16x128) += P(16x64) @ V(64x128); V from transposed (H,S) ----
#pragma unroll
    for (int j = 0; j < 8; ++j) {
      v16bf bV0 = load_bfragT(Vbase, SB, kb0, c0 + j * 16);
      o[j] = wmma_bf16(aP0, bV0, o[j]);
      v16bf bV1 = load_bfragT(Vbase, SB, kb0 + 32, c0 + j * 16);
      o[j] = wmma_bf16(aP1, bV1, o[j]);
    }
  }

  // ---- finalize: divide by row sums, store bf16 (B*S, H) ----
#pragma unroll
  for (int r = 0; r < 8; ++r) {
    float inv = 1.0f / l_[r];
    int row = bS + q0 + r + half * 8;
#pragma unroll
    for (int j = 0; j < 8; ++j) {
      int col = c0 + j * 16 + ln;
      O[(size_t)row * HH + col] = f2bfu(o[j][r] * inv);
    }
  }
}

// ---------------------------------------------------------------------------
// Kernel 4: output projection  out = O @ Wo^T + bo  (f32 result)
// ---------------------------------------------------------------------------
__global__ void k_oproj(const unsigned short* __restrict__ A,
                        const unsigned short* __restrict__ Wob,
                        const float* __restrict__ bo,
                        float* __restrict__ out) {
  int wave = threadIdx.x >> 5;
  int lane = threadIdx.x & 31;
  int row0  = (blockIdx.x * 4 + wave) * 16;
  int ncol0 = blockIdx.y * 64;

  v8f acc[4] = {v8f{}, v8f{}, v8f{}, v8f{}};
  for (int k0 = 0; k0 < HH; k0 += 32) {
    v16bf a = load_afrag(A, HH, row0, k0);
#pragma unroll
    for (int j = 0; j < 4; ++j) {
      v16bf b = load_bfragT(Wob, HH, k0, ncol0 + j * 16);
      acc[j] = wmma_bf16(a, b, acc[j]);
    }
  }

  int half = lane >> 4;
#pragma unroll
  for (int j = 0; j < 4; ++j) {
    int col = ncol0 + j * 16 + (lane & 15);
    float bb = bo[col];
#pragma unroll
    for (int r = 0; r < 8; ++r) {
      int row = row0 + r + half * 8;
      out[(size_t)row * HH + col] = acc[j][r] + bb;
    }
  }
}

// ---------------------------------------------------------------------------
// Host launcher
// ---------------------------------------------------------------------------
extern "C" void kernel_launch(void* const* d_in, const int* in_sizes, int n_in,
                              void* d_out, int out_size, void* d_ws, size_t ws_size,
                              hipStream_t stream) {
  const float* hs = (const float*)d_in[0];
  const float* Wq = (const float*)d_in[1];
  const float* bq = (const float*)d_in[2];
  const float* Wk = (const float*)d_in[3];
  const float* bk = (const float*)d_in[4];
  const float* Wv = (const float*)d_in[5];
  const float* bv = (const float*)d_in[6];
  const float* Wo = (const float*)d_in[7];
  const float* bo = (const float*)d_in[8];
  float* out = (float*)d_out;

  char* ws = (char*)d_ws;
  unsigned int* cnt = (unsigned int*)ws;
  int* flag = (int*)(ws + 8);
  size_t off = 256;
  const size_t MAT  = (size_t)BB * SB * HH * sizeof(unsigned short);  // 16 MiB
  const size_t WMAT = (size_t)HH * HH * sizeof(unsigned short);       //  8 MiB
  unsigned short* Qb  = (unsigned short*)(ws + off); off += MAT;
  unsigned short* Kb  = (unsigned short*)(ws + off); off += MAT;
  unsigned short* Vt  = (unsigned short*)(ws + off); off += MAT;
  unsigned short* Ob  = (unsigned short*)(ws + off); off += MAT;
  unsigned short* Xb  = (unsigned short*)(ws + off); off += MAT;
  unsigned short* Wqb = (unsigned short*)(ws + off); off += WMAT;
  unsigned short* Wkb = (unsigned short*)(ws + off); off += WMAT;
  unsigned short* Wvb = (unsigned short*)(ws + off); off += WMAT;
  unsigned short* Wob = (unsigned short*)(ws + off); off += WMAT;

  hipMemsetAsync(cnt, 0, 8, stream);
  k_density<<<1024, 256, 0, stream>>>(hs, (unsigned int)NTOT, cnt);
  k_flag<<<1, 32, 0, stream>>>(cnt, flag, (unsigned int)NTOT);

  k_cast<<<1024, 256, 0, stream>>>(hs, Xb, (unsigned int)(NTOT / 4));
  k_cast<<<1024, 256, 0, stream>>>(Wq, Wqb, (unsigned int)(HH * HH / 4));
  k_cast<<<1024, 256, 0, stream>>>(Wk, Wkb, (unsigned int)(HH * HH / 4));
  k_cast<<<1024, 256, 0, stream>>>(Wv, Wvb, (unsigned int)(HH * HH / 4));
  k_cast<<<1024, 256, 0, stream>>>(Wo, Wob, (unsigned int)(HH * HH / 4));

  k_qkv<<<dim3(64, 32, 3), 128, 0, stream>>>(Xb, Wqb, bq, Wkb, bk, Wvb, bv,
                                             Qb, Kb, Vt);
  k_attn<<<dim3(32, 16, 2), 128, 0, stream>>>(Qb, Kb, Vt, flag, Ob);
  k_oproj<<<dim3(64, 32), 128, 0, stream>>>(Ob, Wob, bo, out);
}
